// VectorQuantizer2_9758165696769
// MI455X (gfx1250) — matrix operand order repaired and checked by reading, not compiled, not run on hardware
//
#include <hip/hip_runtime.h>
#include <hip/hip_bf16.h>
#include <math.h>

// ---------------------------------------------------------------------------
// Types for CDNA5 WMMA
// ---------------------------------------------------------------------------
typedef __attribute__((ext_vector_type(16))) _Float16 v16h;
typedef __attribute__((ext_vector_type(8)))  float    v8f;

#define NB 128          // batch
#define CIN 16          // C = CVAE/2
#define CV 32           // CVAE
#define VOCAB 4096
#define HW 16           // H == W
#define FHAT_N (NB*CIN*HW*HW)   // 524288

// ---------------------------------------------------------------------------
// 8x8 orthonormal DCT-II basis, computed in-register (cheap, avoids HtoD copy)
// ---------------------------------------------------------------------------
__device__ __forceinline__ void dct_basis(float M[8][8]) {
  const float s0 = 0.35355339059327373f;   // sqrt(1/8)
  const float s1 = 0.5f;                   // sqrt(2/8)
  const float pi16 = 0.19634954084936207f; // pi/16
#pragma unroll
  for (int k = 0; k < 8; ++k)
#pragma unroll
    for (int n = 0; n < 8; ++n)
      M[k][n] = (k == 0 ? s0 : s1) * cosf(pi16 * (float)((2 * n + 1) * k));
}

// ---------------------------------------------------------------------------
// Zero-init output (f_hat + loss scalar)
// ---------------------------------------------------------------------------
__global__ void init_zero(float* __restrict__ p, int n) {
  int i = blockIdx.x * blockDim.x + threadIdx.x;
  if (i < n) p[i] = 0.0f;
}

// ---------------------------------------------------------------------------
// Codebook prep: f32 -> f16 copy + precomputed -0.5*||e||^2 per code
// ---------------------------------------------------------------------------
__global__ void prep_emb(const float* __restrict__ emb,
                         _Float16* __restrict__ emb16,
                         float* __restrict__ esqn) {
  int code = blockIdx.x * blockDim.x + threadIdx.x;
  if (code >= VOCAB) return;
  float s = 0.0f;
#pragma unroll
  for (int k = 0; k < CV; ++k) {
    float v = emb[(size_t)code * CV + k];
    s += v * v;
    emb16[(size_t)code * CV + k] = (_Float16)v;
  }
  esqn[code] = -0.5f * s;
}

// ---------------------------------------------------------------------------
// Forward DCT of all 8x8 blocks of f:  dctb[blk][64] = M * B * M^T
// block id i = ((b*16 + c)*2 + by)*2 + bx
// ---------------------------------------------------------------------------
__global__ void dct_fwd(const float* __restrict__ f, float* __restrict__ dctb) {
  int i = blockIdx.x * blockDim.x + threadIdx.x;
  if (i >= NB * CIN * 2 * 2) return;
  int bx = i & 1, by = (i >> 1) & 1, c = (i >> 2) & 15, bb = i >> 6;
  float M[8][8];
  dct_basis(M);
  float blk[8][8];
#pragma unroll
  for (int y = 0; y < 8; ++y)
#pragma unroll
    for (int x = 0; x < 8; ++x)
      blk[y][x] = f[(((size_t)bb * CIN + c) * HW + by * 8 + y) * HW + bx * 8 + x];
  float t[8][8];
#pragma unroll
  for (int r = 0; r < 8; ++r)
#pragma unroll
    for (int q = 0; q < 8; ++q) {
      float s = 0.0f;
#pragma unroll
      for (int k = 0; k < 8; ++k) s += M[r][k] * blk[k][q];
      t[r][q] = s;
    }
  float* o = dctb + (size_t)i * 64;
#pragma unroll
  for (int r = 0; r < 8; ++r)
#pragma unroll
    for (int q = 0; q < 8; ++q) {
      float s = 0.0f;
#pragma unroll
      for (int k = 0; k < 8; ++k) s += t[r][k] * M[q][k];
      o[r * 8 + q] = s;
    }
}

// ---------------------------------------------------------------------------
// Masked inverse DCT for band si (keep coeffs with max(i,j)==si), write NCHW x
// ---------------------------------------------------------------------------
__global__ void idct_band(const float* __restrict__ dctb, float* __restrict__ x,
                          int si) {
  int i = blockIdx.x * blockDim.x + threadIdx.x;
  if (i >= NB * CIN * 2 * 2) return;
  int bx = i & 1, by = (i >> 1) & 1, c = (i >> 2) & 15, bb = i >> 6;
  float M[8][8];
  dct_basis(M);
  const float* D = dctb + (size_t)i * 64;
  float out[8][8];
#pragma unroll
  for (int y = 0; y < 8; ++y)
#pragma unroll
    for (int xx = 0; xx < 8; ++xx) out[y][xx] = 0.0f;
  for (int r = 0; r < 8; ++r)
    for (int q = 0; q < 8; ++q) {
      int mx = r > q ? r : q;
      if (mx != si) continue;
      float d = D[r * 8 + q];
#pragma unroll
      for (int n = 0; n < 8; ++n) {
        float mn = M[r][n] * d;
#pragma unroll
        for (int m2 = 0; m2 < 8; ++m2) out[n][m2] += mn * M[q][m2];
      }
    }
#pragma unroll
  for (int y = 0; y < 8; ++y)
#pragma unroll
    for (int xx = 0; xx < 8; ++xx)
      x[(((size_t)bb * CIN + c) * HW + by * 8 + y) * HW + bx * 8 + xx] = out[y][xx];
}

// ---------------------------------------------------------------------------
// Generic direct conv2d (+optional SiLU), NCHW, OIHW weights
// ---------------------------------------------------------------------------
__global__ void conv2d_g(const float* __restrict__ in, const float* __restrict__ w,
                         const float* __restrict__ bias, float* __restrict__ out,
                         int Cin, int Cout, int Hin, int Win, int Hout, int Wout,
                         int K, int S, int P, int do_silu) {
  int i = blockIdx.x * blockDim.x + threadIdx.x;
  int total = NB * Cout * Hout * Wout;
  if (i >= total) return;
  int ox = i % Wout;
  int t = i / Wout;
  int oy = t % Hout; t /= Hout;
  int co = t % Cout;
  int bb = t / Cout;
  float acc = bias[co];
  for (int ci = 0; ci < Cin; ++ci)
    for (int ky = 0; ky < K; ++ky) {
      int iy = oy * S - P + ky;
      if ((unsigned)iy >= (unsigned)Hin) continue;
      for (int kx = 0; kx < K; ++kx) {
        int ix = ox * S - P + kx;
        if ((unsigned)ix >= (unsigned)Win) continue;
        acc += in[(((size_t)bb * Cin + ci) * Hin + iy) * Win + ix] *
               w[(((size_t)co * Cin + ci) * K + ky) * K + kx];
      }
    }
  if (do_silu) acc = acc / (1.0f + expf(-acc));
  out[i] = acc;
}

// ---------------------------------------------------------------------------
// Generic transposed conv2d (+SiLU), weights (Cin,Cout,K,K) like torch
// ---------------------------------------------------------------------------
__global__ void convt2d_g(const float* __restrict__ in, const float* __restrict__ w,
                          const float* __restrict__ bias, float* __restrict__ out,
                          int Cin, int Cout, int Hin, int Win, int Hout, int Wout,
                          int K, int S, int P, int do_silu) {
  int i = blockIdx.x * blockDim.x + threadIdx.x;
  int total = NB * Cout * Hout * Wout;
  if (i >= total) return;
  int ox = i % Wout;
  int t = i / Wout;
  int oy = t % Hout; t /= Hout;
  int co = t % Cout;
  int bb = t / Cout;
  float acc = bias[co];
  for (int ci = 0; ci < Cin; ++ci)
    for (int ky = 0; ky < K; ++ky) {
      int ty = oy + P - ky;
      if (ty < 0 || (ty % S) != 0) continue;
      int iy = ty / S;
      if (iy >= Hin) continue;
      for (int kx = 0; kx < K; ++kx) {
        int tx = ox + P - kx;
        if (tx < 0 || (tx % S) != 0) continue;
        int ix = tx / S;
        if (ix >= Win) continue;
        acc += in[(((size_t)bb * Cin + ci) * Hin + iy) * Win + ix] *
               w[(((size_t)ci * Cout + co) * K + ky) * K + kx];
      }
    }
  if (do_silu) acc = acc / (1.0f + expf(-acc));
  out[i] = acc;
}

// ---------------------------------------------------------------------------
// VQ argmin over 4096 codes via WMMA f32_16x16x32_f16.
//   score = z.e - 0.5*||e||^2   (C-initializer carries esqn = -0.5*||e||^2)
//   argmin(dist) == argmax(score); tie-break -> lowest index (matches argmin)
// Each wave owns a 32-point group = TWO A tiles -> 2 WMMAs per B-load.
// z is encoder output NCHW: channel k of point pt at z[b*32*pn2 + k*pn2 + rem]
// ---------------------------------------------------------------------------
__global__ void __launch_bounds__(128)
vq_argmin_wmma(const float* __restrict__ z, const _Float16* __restrict__ emb16,
               const float* __restrict__ esqn, int* __restrict__ idx,
               int ngroups, int pn2) {
  int wave = threadIdx.x >> 5;
  int grp = blockIdx.x * 4 + wave;
  if (grp >= ngroups) return;
  int lane = threadIdx.x & 31;
  int m = lane & 15;
  int hi = lane >> 4;

  // ---- Two A matrices (16 points x 32 dims each, f16), contiguous k order, same as B:
  //      lane m holds K = 16*hi..+16 of its row.
  v16h a0, a1;
#pragma unroll
  for (int tt = 0; tt < 2; ++tt) {
    int pt = grp * 32 + tt * 16 + m;
    int bb = pt / pn2;
    int rem = pt - bb * pn2;
    const float* zr = z + (size_t)bb * CV * pn2 + rem;  // channel k at zr[k*pn2]
    v16h a;
#pragma unroll
    for (int j = 0; j < 16; ++j)
      a[j] = (_Float16)zr[(size_t)(16 * hi + j) * pn2];
    if (tt == 0) a0 = a; else a1 = a;
  }

  float best0[8], best1[8];
  int bidx0[8], bidx1[8];
#pragma unroll
  for (int j = 0; j < 8; ++j) {
    best0[j] = -3.4e38f; bidx0[j] = 0;
    best1[j] = -3.4e38f; bidx1[j] = 0;
  }

#pragma unroll 2
  for (int ct = 0; ct < VOCAB / 16; ++ct) {
    int code = ct * 16 + m;  // B column n == lane%16; D column == lane%16 too
    // B matrix (32 x 16 codes): lane n holds K = 16*hi..+16 of column n
    const v16h* bp = (const v16h*)(emb16 + ((size_t)code * CV + 16 * hi));
    v16h bv = *bp;
    __builtin_prefetch(emb16 + ((size_t)((code + 16) & (VOCAB - 1)) * CV), 0, 1);
    float cinit = esqn[code];
    v8f c;
#pragma unroll
    for (int j = 0; j < 8; ++j) c[j] = cinit;
    // Two back-to-back WMMAs sharing B: D = A x B + (-0.5*||e||^2)
    v8f d0 = __builtin_amdgcn_wmma_f32_16x16x32_f16(
        false, a0, false, bv, (short)0, c, false, false);
    v8f d1 = __builtin_amdgcn_wmma_f32_16x16x32_f16(
        false, a1, false, bv, (short)0, c, false, false);
#pragma unroll
    for (int j = 0; j < 8; ++j)
      if (d0[j] > best0[j]) { best0[j] = d0[j]; bidx0[j] = code; }
#pragma unroll
    for (int j = 0; j < 8; ++j)
      if (d1[j] > best1[j]) { best1[j] = d1[j]; bidx1[j] = code; }
  }

  // Per-row argmax: row (j + 8*hi) lives in the 16 lanes of this half.
#pragma unroll
  for (int j = 0; j < 8; ++j) {
#pragma unroll
    for (int xm = 1; xm < 16; xm <<= 1) {
      float ov0 = __shfl_xor(best0[j], xm, 32);
      int oi0 = __shfl_xor(bidx0[j], xm, 32);
      if (ov0 > best0[j] || (ov0 == best0[j] && oi0 < bidx0[j])) {
        best0[j] = ov0; bidx0[j] = oi0;
      }
      float ov1 = __shfl_xor(best1[j], xm, 32);
      int oi1 = __shfl_xor(bidx1[j], xm, 32);
      if (ov1 > best1[j] || (ov1 == best1[j] && oi1 < bidx1[j])) {
        best1[j] = ov1; bidx1[j] = oi1;
      }
    }
    if (m == 0) {
      idx[grp * 32 + 8 * hi + j] = bidx0[j];
      idx[grp * 32 + 16 + 8 * hi + j] = bidx1[j];
    }
  }
}

// ---------------------------------------------------------------------------
// hq0[b, c, py, px] = emb[idx[(b*pn+py)*pn+px]][c]
// ---------------------------------------------------------------------------
__global__ void gather_hq(const float* __restrict__ emb, const int* __restrict__ idx,
                          float* __restrict__ hq, int npts, int pn) {
  int i = blockIdx.x * blockDim.x + threadIdx.x;
  if (i >= npts * CV) return;
  int c = i & 31;
  int p = i >> 5;
  int px = p % pn;
  int t = p / pn;
  int py = t % pn;
  int bb = t / pn;
  hq[(((size_t)bb * CV + c) * pn + py) * pn + px] =
      emb[(size_t)idx[p] * CV + c];
}

// ---------------------------------------------------------------------------
// Phi smoothing: out = 0.5*hq + 0.5*(conv3x3(hq) + b), 16 channels, 16x16
// ---------------------------------------------------------------------------
__global__ void phi_blend(const float* __restrict__ hq, const float* __restrict__ w,
                          const float* __restrict__ bias, float* __restrict__ out) {
  int i = blockIdx.x * blockDim.x + threadIdx.x;
  if (i >= FHAT_N) return;
  int ox = i % HW;
  int t = i / HW;
  int oy = t % HW; t /= HW;
  int co = t % CIN;
  int bb = t / CIN;
  float acc = bias[co];
  for (int ci = 0; ci < CIN; ++ci)
#pragma unroll
    for (int ky = 0; ky < 3; ++ky) {
      int iy = oy - 1 + ky;
      if ((unsigned)iy >= (unsigned)HW) continue;
#pragma unroll
      for (int kx = 0; kx < 3; ++kx) {
        int ix = ox - 1 + kx;
        if ((unsigned)ix >= (unsigned)HW) continue;
        acc += hq[(((size_t)bb * CIN + ci) * HW + iy) * HW + ix] *
               w[(((size_t)co * CIN + ci) * 3 + ky) * 3 + kx];
      }
    }
  out[i] = 0.5f * hq[i] + 0.5f * acc;
}

// ---------------------------------------------------------------------------
// f_hat += hq; loss += scale * sum((f_hat - f)^2)   (block-reduced atomicAdd)
// ---------------------------------------------------------------------------
__global__ void accum_loss(float* __restrict__ fhat, const float* __restrict__ hq,
                           const float* __restrict__ f, float* __restrict__ loss,
                           float scale, int n) {
  int i = blockIdx.x * blockDim.x + threadIdx.x;
  float d = 0.0f;
  if (i < n) {
    float v = fhat[i] + hq[i];
    fhat[i] = v;
    float df = v - f[i];
    d = df * df;
  }
#pragma unroll
  for (int o = 16; o > 0; o >>= 1) d += __shfl_down(d, o, 32);
  __shared__ float sm[8];
  int lane = threadIdx.x & 31, wv = threadIdx.x >> 5;
  if (lane == 0) sm[wv] = d;
  __syncthreads();
  if (wv == 0) {
    d = (lane < 8) ? sm[lane] : 0.0f;
#pragma unroll
    for (int o = 4; o > 0; o >>= 1) d += __shfl_down(d, o, 32);
    if (lane == 0) atomicAdd(loss, d * scale);
  }
}

// ---------------------------------------------------------------------------
// Host-side model description (mirrors CONV_SPECS / DECONV_SPECS)
// ---------------------------------------------------------------------------
struct CSpec { int ci, co, k, s, p; };
struct DSpec { int ci, co, k, s, p, t; };

static const int NCONV[8] = {4, 3, 2, 1, 1, 1, 1, 1};
static const CSpec CONV[8][4] = {
    {{16,32,3,2,1},{32,32,3,2,1},{32,32,3,2,1},{32,32,2,1,0}},
    {{16,32,3,2,1},{32,32,3,2,1},{32,32,3,2,1},{0,0,0,0,0}},
    {{16,32,3,2,1},{32,32,3,2,1},{0,0,0,0,0},{0,0,0,0,0}},
    {{16,32,5,2,0},{0,0,0,0,0},{0,0,0,0,0},{0,0,0,0,0}},
    {{16,32,3,2,1},{0,0,0,0,0},{0,0,0,0,0},{0,0,0,0,0}},
    {{16,32,7,1,0},{0,0,0,0,0},{0,0,0,0,0},{0,0,0,0,0}},
    {{16,32,4,1,0},{0,0,0,0,0},{0,0,0,0,0},{0,0,0,0,0}},
    {{16,32,3,1,1},{0,0,0,0,0},{0,0,0,0,0},{0,0,0,0,0}},
};
static const int NDEC[8] = {4, 3, 2, 1, 1, 1, 1, 1};
static const DSpec DEC[8][4] = {
    {{32,32,2,1,0,1},{32,32,4,2,1,1},{32,32,4,2,1,1},{32,16,4,2,1,1}},
    {{32,32,4,2,1,1},{32,32,4,2,1,1},{32,16,4,2,1,1},{0,0,0,0,0,0}},
    {{32,32,4,2,1,1},{32,16,4,2,1,1},{0,0,0,0,0,0},{0,0,0,0,0,0}},
    {{32,16,6,2,0,1},{0,0,0,0,0,0},{0,0,0,0,0,0},{0,0,0,0,0,0}},
    {{32,16,4,2,1,1},{0,0,0,0,0,0},{0,0,0,0,0,0},{0,0,0,0,0,0}},
    {{32,16,7,1,0,1},{0,0,0,0,0,0},{0,0,0,0,0,0},{0,0,0,0,0,0}},
    {{32,16,4,1,0,1},{0,0,0,0,0,0},{0,0,0,0,0,0},{0,0,0,0,0,0}},
    {{32,16,3,1,1,0},{0,0,0,0,0,0},{0,0,0,0,0,0},{0,0,0,0,0,0}},
};
static const int PNS[8] = {1, 2, 4, 6, 8, 10, 13, 16};
static const int PHIM[8] = {0, 0, 1, 1, 2, 2, 3, 3};

// Workspace layout (bytes)
static const size_t OFF_EMB16 = 0;                         // 256 KB (f16 codebook)
static const size_t OFF_ESQ   = 256 * 1024;                // 16 KB (-0.5*||e||^2)
static const size_t OFF_DCT   = 512 * 1024;                // 2 MB
static const size_t OFF_X     = OFF_DCT + 2 * 1024 * 1024; // 2 MB
static const size_t OFF_A     = OFF_X + 2 * 1024 * 1024;   // 4 MB (conv ping)
static const size_t OFF_B     = OFF_A + 4 * 1024 * 1024;   // 4 MB (conv pong)
static const size_t OFF_IDX   = OFF_B + 4 * 1024 * 1024;   // 128 KB
static const size_t OFF_HQA   = OFF_IDX + 1024 * 1024;     // 4 MB
static const size_t OFF_HQB   = OFF_HQA + 4 * 1024 * 1024; // 4 MB

extern "C" void kernel_launch(void* const* d_in, const int* in_sizes, int n_in,
                              void* d_out, int out_size, void* d_ws, size_t ws_size,
                              hipStream_t stream) {
  (void)in_sizes; (void)n_in; (void)out_size; (void)ws_size;

  const float* f = (const float*)d_in[0];
  const float* emb = (const float*)d_in[1];

  // flatten param pointers in setup_inputs() order
  int pi = 2;
  const float *cw[8][4], *cb[8][4], *dw[8][4], *db[8][4], *pw[4], *pb[4];
  for (int si = 0; si < 8; ++si)
    for (int l = 0; l < NCONV[si]; ++l) {
      cw[si][l] = (const float*)d_in[pi++];
      cb[si][l] = (const float*)d_in[pi++];
    }
  for (int si = 0; si < 8; ++si)
    for (int l = 0; l < NDEC[si]; ++l) {
      dw[si][l] = (const float*)d_in[pi++];
      db[si][l] = (const float*)d_in[pi++];
    }
  for (int q = 0; q < 4; ++q) {
    pw[q] = (const float*)d_in[pi++];
    pb[q] = (const float*)d_in[pi++];
  }

  char* ws = (char*)d_ws;
  _Float16* emb16 = (_Float16*)(ws + OFF_EMB16);
  float* esqn = (float*)(ws + OFF_ESQ);
  float* dctb = (float*)(ws + OFF_DCT);
  float* xb   = (float*)(ws + OFF_X);
  float* bufA = (float*)(ws + OFF_A);
  float* bufB = (float*)(ws + OFF_B);
  int*   idxb = (int*)(ws + OFF_IDX);
  float* hqA  = (float*)(ws + OFF_HQA);
  float* hqB  = (float*)(ws + OFF_HQB);

  float* fhat = (float*)d_out;              // FHAT_N floats
  float* loss = fhat + FHAT_N;              // 1 float

  init_zero<<<(FHAT_N + 1 + 255) / 256, 256, 0, stream>>>(fhat, FHAT_N + 1);
  prep_emb<<<VOCAB / 256, 256, 0, stream>>>(emb, emb16, esqn);
  dct_fwd<<<(NB * CIN * 4 + 255) / 256, 256, 0, stream>>>(f, dctb);

  const float loss_scale = 2.0f / (8.0f * (float)FHAT_N);

  for (int si = 0; si < 8; ++si) {
    idct_band<<<(NB * CIN * 4 + 255) / 256, 256, 0, stream>>>(dctb, xb, si);

    // ---- encoder ----
    const float* cur = xb;
    int Hc = HW;
    for (int l = 0; l < NCONV[si]; ++l) {
      CSpec s = CONV[si][l];
      int Ho = (Hc + 2 * s.p - s.k) / s.s + 1;
      float* ob = (l & 1) ? bufB : bufA;
      int total = NB * s.co * Ho * Ho;
      conv2d_g<<<(total + 255) / 256, 256, 0, stream>>>(
          cur, cw[si][l], cb[si][l], ob, s.ci, s.co, Hc, Hc, Ho, Ho,
          s.k, s.s, s.p, 1);
      cur = ob;
      Hc = Ho;
    }

    // ---- vector quantization (WMMA) ----
    int pn = PNS[si];
    int pn2 = pn * pn;
    int npts = NB * pn2;
    int ngroups = npts / 32;   // 32 points (2 A tiles) per wave
    vq_argmin_wmma<<<(ngroups + 3) / 4, 128, 0, stream>>>(
        cur, emb16, esqn, idxb, ngroups, pn2);
    gather_hq<<<(npts * CV + 255) / 256, 256, 0, stream>>>(
        emb, idxb, hqA, npts, pn);

    // ---- decoder ----
    const float* dcur = hqA;
    Hc = pn;
    for (int l = 0; l < NDEC[si]; ++l) {
      DSpec s = DEC[si][l];
      int Ho = s.t ? ((Hc - 1) * s.s - 2 * s.p + s.k)
                   : ((Hc + 2 * s.p - s.k) / s.s + 1);
      float* ob = (dcur == hqA) ? hqB : hqA;
      int total = NB * s.co * Ho * Ho;
      if (s.t)
        convt2d_g<<<(total + 255) / 256, 256, 0, stream>>>(
            dcur, dw[si][l], db[si][l], ob, s.ci, s.co, Hc, Hc, Ho, Ho,
            s.k, s.s, s.p, 1);
      else
        conv2d_g<<<(total + 255) / 256, 256, 0, stream>>>(
            dcur, dw[si][l], db[si][l], ob, s.ci, s.co, Hc, Hc, Ho, Ho,
            s.k, s.s, s.p, 1);
      dcur = ob;
      Hc = Ho;
    }

    // ---- phi residual smoothing ----
    float* pout = (dcur == hqA) ? hqB : hqA;
    phi_blend<<<(FHAT_N + 255) / 256, 256, 0, stream>>>(
        dcur, pw[PHIM[si]], pb[PHIM[si]], pout);

    // ---- accumulate f_hat and loss ----
    accum_loss<<<(FHAT_N + 255) / 256, 256, 0, stream>>>(
        fhat, pout, f, loss, loss_scale, FHAT_N);
  }
}
